// MixtralMegaBlocksAdapter_47029891891651
// MI455X (gfx1250) — matrix operand order repaired and checked by reading, not compiled, not run on hardware
//
#include <hip/hip_runtime.h>

// ---------------- problem constants ----------------
#define B_ 4
#define S_ 2048
#define H_ 1024
#define F_ 3584
#define E_ 8
#define K_ 2
#define T_ (B_ * S_)          // 8192 tokens

// ---------------- tiling ----------------
#define BM 64                  // rows per block
#define BN 128                 // cols per block
#define BK 32                  // K step (bf16 wmma K)
#define LDS_STRIDE 40          // 32 + 8 pad elems -> 80B row stride (16B aligned, bank-spread)

typedef __bf16 bf16;
typedef __attribute__((ext_vector_type(16))) __bf16 v16bf;
typedef __attribute__((ext_vector_type(8)))  __bf16 v8bf;
typedef __attribute__((ext_vector_type(4)))  __bf16 v4bf;
typedef __attribute__((ext_vector_type(8)))  float  v8f;
typedef __attribute__((ext_vector_type(4)))  int    v4i;

// ---------------------------------------------------------------------------
// CDNA5 async global->LDS copy (ASYNCcnt-tracked), with sync b128 fallback.
// Builtin signature (from clang diagnostic): (v4i AS1*, v4i AS3*, Ii, Ii)
// ---------------------------------------------------------------------------
#if defined(__has_builtin)
#if __has_builtin(__builtin_amdgcn_global_load_async_to_lds_b128)
#define HAVE_ASYNC_LDS 1
#endif
#endif

typedef __attribute__((address_space(1))) v4i* as1_v4i_ptr;
typedef __attribute__((address_space(3))) v4i* as3_v4i_ptr;

#if defined(HAVE_ASYNC_LDS)
__device__ __forceinline__ void async_cp16(void* lds, const void* g) {
  __builtin_amdgcn_global_load_async_to_lds_b128(
      (as1_v4i_ptr)(const_cast<void*>(g)), (as3_v4i_ptr)(lds), 0, 0);
}
#if __has_builtin(__builtin_amdgcn_s_wait_asynccnt)
#define ASYNC_WAIT(n) __builtin_amdgcn_s_wait_asynccnt(n)
#else
#define ASYNC_WAIT(n) asm volatile("s_wait_asynccnt %0" ::"i"(n) : "memory")
#endif
#else
__device__ __forceinline__ void async_cp16(void* lds, const void* g) {
  *(v8bf*)lds = *(const v8bf*)g;
}
#define ASYNC_WAIT(n) do { } while (0)
#endif

// ---------------------------------------------------------------------------
// WMMA fragment loaders from LDS (bf16, 16x16x32).  Two ds_load_b128 each.
// ---------------------------------------------------------------------------
__device__ __forceinline__ v16bf load_frag_a(const bf16* rowbase, int lane) {
  int r = lane & 15, h = lane >> 4;
  const bf16* p = rowbase + r * LDS_STRIDE;
  v8bf lo = *(const v8bf*)(p + 8 * h);
  v8bf hi = *(const v8bf*)(p + 16 + 8 * h);
  v16bf o;
#pragma unroll
  for (int i = 0; i < 8; i++) { o[i] = lo[i]; o[i + 8] = hi[i]; }
  return o;
}

__device__ __forceinline__ v16bf load_frag_b(const bf16* rowbase, int lane) {
  int n = lane & 15, h = lane >> 4;
  const bf16* p = rowbase + n * LDS_STRIDE;
  v8bf lo = *(const v8bf*)(p + 8 * h);
  v8bf hi = *(const v8bf*)(p + 16 + 8 * h);
  v16bf o;
#pragma unroll
  for (int i = 0; i < 8; i++) { o[i] = lo[i]; o[i + 8] = hi[i]; }
  return o;
}

__device__ __forceinline__ v8f wmma_bf16(v16bf a, v16bf b, v8f c) {
  return __builtin_amdgcn_wmma_f32_16x16x32_bf16(false, a, false, b, (short)0, c,
                                                 false, false);
}

// ---------------------------------------------------------------------------
// Kernel: router (one wave32 per token) -> expert lists via atomic slots.
// ---------------------------------------------------------------------------
__global__ __launch_bounds__(256) void router_kernel(
    const float* __restrict__ x, const float* __restrict__ rw,
    int* __restrict__ counts, int* __restrict__ tok, float* __restrict__ wgt) {
  int wave = (int)((blockIdx.x * blockDim.x + threadIdx.x) >> 5);
  int lane = threadIdx.x & 31;
  if (wave >= T_) return;
  const float* xp = x + (size_t)wave * H_;

  float acc[E_];
#pragma unroll
  for (int e = 0; e < E_; e++) acc[e] = 0.f;
  for (int h = lane; h < H_; h += 32) {
    float xv = xp[h];
#pragma unroll
    for (int e = 0; e < E_; e++) acc[e] += xv * rw[e * H_ + h];
  }
#pragma unroll
  for (int e = 0; e < E_; e++) {
#pragma unroll
    for (int s = 16; s > 0; s >>= 1) acc[e] += __shfl_xor(acc[e], s, 32);
  }
  if (lane == 0) {
    float m = acc[0];
#pragma unroll
    for (int e = 1; e < E_; e++) m = fmaxf(m, acc[e]);
    float p[E_];
#pragma unroll
    for (int e = 0; e < E_; e++) p[e] = __expf(acc[e] - m);
    int i0 = 0;
#pragma unroll
    for (int e = 1; e < E_; e++) if (p[e] > p[i0]) i0 = e;
    int i1 = (i0 == 0) ? 1 : 0;
#pragma unroll
    for (int e = 0; e < E_; e++) if (e != i0 && p[e] > p[i1]) i1 = e;
    float inv = 1.f / (p[i0] + p[i1]);
    int s0 = atomicAdd(&counts[i0], 1);
    tok[i0 * T_ + s0] = (wave << 1) | 0;
    wgt[i0 * T_ + s0] = p[i0] * inv;
    int s1 = atomicAdd(&counts[i1], 1);
    tok[i1 * T_ + s1] = (wave << 1) | 1;
    wgt[i1 * T_ + s1] = p[i1] * inv;
  }
}

// ---------------------------------------------------------------------------
// Kernel: E=8 exclusive scan.
// ---------------------------------------------------------------------------
__global__ void scan_kernel(const int* __restrict__ counts, int* __restrict__ offsets) {
  if (threadIdx.x == 0) {
    int a = 0;
#pragma unroll
    for (int e = 0; e < E_; e++) { offsets[e] = a; a += counts[e]; }
    offsets[E_] = a;
  }
}

// ---------------------------------------------------------------------------
// Kernel: f32 -> bf16 bulk convert (float4 in, 4x bf16 out).
// ---------------------------------------------------------------------------
__global__ __launch_bounds__(256) void cvt_kernel(const float4* __restrict__ s,
                                                  v4bf* __restrict__ d) {
  size_t i = (size_t)blockIdx.x * blockDim.x + threadIdx.x;
  float4 f = s[i];
  v4bf o;
  o[0] = (bf16)f.x; o[1] = (bf16)f.y; o[2] = (bf16)f.z; o[3] = (bf16)f.w;
  d[i] = o;
}

// ---------------------------------------------------------------------------
// Kernel: w2 [E,F,H] f32 -> w2t [E,H,F] bf16 (LDS-tiled transpose).
// ---------------------------------------------------------------------------
__global__ void w2t_kernel(const float* __restrict__ w2, bf16* __restrict__ w2t) {
  __shared__ float tile[32][33];
  int e = blockIdx.z;
  const float* src = w2 + (size_t)e * F_ * H_;
  bf16* dst = w2t + (size_t)e * F_ * H_;
  int n = blockIdx.x * 32 + threadIdx.x;   // H index (coalesced read)
  int k = blockIdx.y * 32 + threadIdx.y;   // F index
#pragma unroll
  for (int j = 0; j < 32; j += 8)
    tile[threadIdx.y + j][threadIdx.x] = src[(size_t)(k + j) * H_ + n];
  __syncthreads();
  int k2 = blockIdx.y * 32 + threadIdx.x;  // F index (coalesced write)
  int n2 = blockIdx.x * 32 + threadIdx.y;  // H index
#pragma unroll
  for (int j = 0; j < 32; j += 8)
    dst[(size_t)(n2 + j) * F_ + k2] = (bf16)tile[threadIdx.x][threadIdx.y + j];
}

// ---------------------------------------------------------------------------
// Kernel: GLU up-projection, double-buffered async tiles.
// hmid[row,n] = silu(X w1^T) * (X v1^T)   (all operands bf16 in ws)
// ---------------------------------------------------------------------------
__global__ __launch_bounds__(256) void glu_kernel(
    const bf16* __restrict__ xb, const bf16* __restrict__ w1b,
    const bf16* __restrict__ v1b, const int* __restrict__ counts,
    const int* __restrict__ offsets, const int* __restrict__ tok,
    bf16* __restrict__ hmid) {
  int e = blockIdx.z;
  int ne = counts[e];
  int m0 = blockIdx.x * BM;
  if (m0 >= ne) return;
  int n0 = blockIdx.y * BN;
  const bf16* W1 = w1b + (size_t)e * F_ * H_;
  const bf16* V1 = v1b + (size_t)e * F_ * H_;

  __shared__ __align__(16) bf16 Xs[2][BM * LDS_STRIDE];
  __shared__ __align__(16) bf16 W1s[2][BN * LDS_STRIDE];
  __shared__ __align__(16) bf16 V1s[2][BN * LDS_STRIDE];
  __shared__ int rows_s[BM];

  int tid = threadIdx.x;
  if (tid < BM) {
    int mi = m0 + tid;
    rows_s[tid] = (mi < ne) ? (tok[e * T_ + mi] >> 1) : 0;
  }
  __syncthreads();

  // per-thread 16B copy chunks: X (1), W1 (2), V1 (2)
  int cr = tid >> 2, cc = tid & 3;
  const bf16* xsrc  = xb + (size_t)rows_s[cr] * H_ + cc * 8;
  const bf16* w1s0  = W1 + (size_t)(n0 + cr) * H_ + cc * 8;
  const bf16* w1s1  = W1 + (size_t)(n0 + cr + 64) * H_ + cc * 8;
  const bf16* v1s0  = V1 + (size_t)(n0 + cr) * H_ + cc * 8;
  const bf16* v1s1  = V1 + (size_t)(n0 + cr + 64) * H_ + cc * 8;
  int dst0 = cr * LDS_STRIDE + cc * 8;
  int dst1 = (cr + 64) * LDS_STRIDE + cc * 8;

  auto issue = [&](int k0, int p) {
    async_cp16(&Xs[p][dst0],  xsrc + k0);
    async_cp16(&W1s[p][dst0], w1s0 + k0);
    async_cp16(&W1s[p][dst1], w1s1 + k0);
    async_cp16(&V1s[p][dst0], v1s0 + k0);
    async_cp16(&V1s[p][dst1], v1s1 + k0);
  };

  int lane = tid & 31, wv = tid >> 5;
  int wm = wv & 3, wn = wv >> 2;               // 4 x 2 wave grid

  v8f accU[4], accV[4];
#pragma unroll
  for (int j = 0; j < 4; j++) { accU[j] = (v8f)(0.0f); accV[j] = (v8f)(0.0f); }

  issue(0, 0);
  for (int k0 = 0, it = 0; k0 < H_; k0 += BK, ++it) {
    int p = it & 1;
    if (k0 + BK < H_) { issue(k0 + BK, p ^ 1); ASYNC_WAIT(5); }
    else              { ASYNC_WAIT(0); }
    __syncthreads();

    v16bf a = load_frag_a(&Xs[p][(wm * 16) * LDS_STRIDE], lane);
#pragma unroll
    for (int j = 0; j < 4; j++) {
      v16bf bw = load_frag_b(&W1s[p][(wn * 64 + j * 16) * LDS_STRIDE], lane);
      accU[j] = wmma_bf16(a, bw, accU[j]);
      v16bf bv = load_frag_b(&V1s[p][(wn * 64 + j * 16) * LDS_STRIDE], lane);
      accV[j] = wmma_bf16(a, bv, accV[j]);
    }
    __syncthreads();
  }

  int base = offsets[e];
#pragma unroll
  for (int j = 0; j < 4; j++) {
    int n = n0 + wn * 64 + j * 16 + (lane & 15);
#pragma unroll
    for (int r = 0; r < 8; r++) {
      int m = wm * 16 + 8 * (lane >> 4) + r;
      int mi = m0 + m;
      if (mi < ne) {
        float u = accU[j][r], v = accV[j][r];
        float hm = __fdividef(u, 1.f + __expf(-u)) * v;
        hmid[(size_t)(base + mi) * F_ + n] = (bf16)hm;
      }
    }
  }
}

// ---------------------------------------------------------------------------
// Kernel: down-projection, double-buffered async tiles.
// part[k][t][h] = w * (hmid w2)   (B = rows of pre-transposed w2t)
// ---------------------------------------------------------------------------
__global__ __launch_bounds__(256) void down_kernel(
    const bf16* __restrict__ w2t, const int* __restrict__ counts,
    const int* __restrict__ offsets, const int* __restrict__ tok,
    const float* __restrict__ wgt, const bf16* __restrict__ hmid,
    float* __restrict__ part) {
  int e = blockIdx.z;
  int ne = counts[e];
  int m0 = blockIdx.x * BM;
  if (m0 >= ne) return;
  int n0 = blockIdx.y * BN;
  const bf16* W2t = w2t + (size_t)e * F_ * H_;
  int base = offsets[e];

  __shared__ __align__(16) bf16 As[2][BM * LDS_STRIDE];
  __shared__ __align__(16) bf16 Bs[2][BN * LDS_STRIDE];
  __shared__ int tk_s[BM];
  __shared__ float wg_s[BM];

  int tid = threadIdx.x;
  if (tid < BM) {
    int mi = m0 + tid;
    tk_s[tid] = (mi < ne) ? tok[e * T_ + mi] : 0;
    wg_s[tid] = (mi < ne) ? wgt[e * T_ + mi] : 0.f;
  }
  __syncthreads();

  // per-thread 16B chunks: A (1), B (2).  Pad-row reads stay inside ws.
  int cr = tid >> 2, cc = tid & 3;
  const bf16* asrc = hmid + (size_t)(base + m0 + cr) * F_ + cc * 8;
  const bf16* bs0  = W2t + (size_t)(n0 + cr) * F_ + cc * 8;
  const bf16* bs1  = W2t + (size_t)(n0 + cr + 64) * F_ + cc * 8;
  int dst0 = cr * LDS_STRIDE + cc * 8;
  int dst1 = (cr + 64) * LDS_STRIDE + cc * 8;

  auto issue = [&](int k0, int p) {
    async_cp16(&As[p][dst0], asrc + k0);
    async_cp16(&Bs[p][dst0], bs0 + k0);
    async_cp16(&Bs[p][dst1], bs1 + k0);
  };

  int lane = tid & 31, wv = tid >> 5;
  int wm = wv & 3, wn = wv >> 2;

  v8f acc[4];
#pragma unroll
  for (int j = 0; j < 4; j++) acc[j] = (v8f)(0.0f);

  issue(0, 0);
  for (int k0 = 0, it = 0; k0 < F_; k0 += BK, ++it) {
    int p = it & 1;
    if (k0 + BK < F_) { issue(k0 + BK, p ^ 1); ASYNC_WAIT(3); }
    else              { ASYNC_WAIT(0); }
    __syncthreads();

    v16bf a = load_frag_a(&As[p][(wm * 16) * LDS_STRIDE], lane);
#pragma unroll
    for (int j = 0; j < 4; j++) {
      v16bf b = load_frag_b(&Bs[p][(wn * 64 + j * 16) * LDS_STRIDE], lane);
      acc[j] = wmma_bf16(a, b, acc[j]);
    }
    __syncthreads();
  }

#pragma unroll
  for (int j = 0; j < 4; j++) {
    int n = n0 + wn * 64 + j * 16 + (lane & 15);
#pragma unroll
    for (int r = 0; r < 8; r++) {
      int m = wm * 16 + 8 * (lane >> 4) + r;
      int mi = m0 + m;
      if (mi < ne) {
        int tk = tk_s[m];
        float v = acc[j][r] * wg_s[m];
        part[((size_t)(tk & 1) * T_ + (size_t)(tk >> 1)) * H_ + n] = v;
      }
    }
  }
}

// ---------------------------------------------------------------------------
// Kernel: out = partial0 + partial1
// ---------------------------------------------------------------------------
__global__ __launch_bounds__(256) void combine_kernel(const float4* __restrict__ p,
                                                      float4* __restrict__ out) {
  size_t i = (size_t)blockIdx.x * blockDim.x + threadIdx.x;
  float4 a = p[i];
  float4 b = p[i + (size_t)T_ * H_ / 4];
  out[i] = make_float4(a.x + b.x, a.y + b.y, a.z + b.z, a.w + b.w);
}

// ---------------------------------------------------------------------------
// workspace layout (bytes)
// ---------------------------------------------------------------------------
static const size_t WS_CNT  = 0;                                    // E ints
static const size_t WS_OFF  = 256;                                  // E+1 ints
static const size_t WS_TOK  = 512;                                  // E*T ints
static const size_t WS_WGT  = WS_TOK + (size_t)E_ * T_ * 4;         // E*T f32
static const size_t WS_XB   = WS_WGT + (size_t)E_ * T_ * 4;         // T*H bf16
static const size_t WS_W1B  = WS_XB + (size_t)T_ * H_ * 2;          // E*F*H bf16
static const size_t WS_V1B  = WS_W1B + (size_t)E_ * F_ * H_ * 2;
static const size_t WS_W2T  = WS_V1B + (size_t)E_ * F_ * H_ * 2;    // transposed
static const size_t WS_HMID = WS_W2T + (size_t)E_ * F_ * H_ * 2;    // T*K x F bf16
static const size_t WS_PART = WS_HMID + (size_t)T_ * K_ * F_ * 2;   // 2*T*H f32

extern "C" void kernel_launch(void* const* d_in, const int* in_sizes, int n_in,
                              void* d_out, int out_size, void* d_ws, size_t ws_size,
                              hipStream_t stream) {
  (void)in_sizes; (void)n_in; (void)out_size; (void)ws_size;
  const float* x  = (const float*)d_in[0];
  const float* rw = (const float*)d_in[1];
  const float* w1 = (const float*)d_in[2];
  const float* v1 = (const float*)d_in[3];
  const float* w2 = (const float*)d_in[4];
  float* out = (float*)d_out;

  char* ws = (char*)d_ws;
  int*   counts  = (int*)(ws + WS_CNT);
  int*   offsets = (int*)(ws + WS_OFF);
  int*   tok     = (int*)(ws + WS_TOK);
  float* wgt     = (float*)(ws + WS_WGT);
  bf16*  xb      = (bf16*)(ws + WS_XB);
  bf16*  w1b     = (bf16*)(ws + WS_W1B);
  bf16*  v1b     = (bf16*)(ws + WS_V1B);
  bf16*  w2t     = (bf16*)(ws + WS_W2T);
  bf16*  hmid    = (bf16*)(ws + WS_HMID);
  float* part    = (float*)(ws + WS_PART);

  (void)hipMemsetAsync(counts, 0, 256, stream);
  router_kernel<<<T_ / 8, 256, 0, stream>>>(x, rw, counts, tok, wgt);
  scan_kernel<<<1, 32, 0, stream>>>(counts, offsets);

  cvt_kernel<<<(T_ * H_) / 1024, 256, 0, stream>>>((const float4*)x, (v4bf*)xb);
  cvt_kernel<<<(E_ * F_ * H_) / 1024, 256, 0, stream>>>((const float4*)w1, (v4bf*)w1b);
  cvt_kernel<<<(E_ * F_ * H_) / 1024, 256, 0, stream>>>((const float4*)v1, (v4bf*)v1b);
  w2t_kernel<<<dim3(H_ / 32, F_ / 32, E_), dim3(32, 8), 0, stream>>>(w2, w2t);

  glu_kernel<<<dim3(T_ / BM, F_ / BN, E_), 256, 0, stream>>>(xb, w1b, v1b, counts,
                                                             offsets, tok, hmid);
  down_kernel<<<dim3(T_ / BM, H_ / BN, E_), 256, 0, stream>>>(w2t, counts, offsets,
                                                              tok, wgt, hmid, part);
  combine_kernel<<<(T_ * (H_ / 4)) / 256, 256, 0, stream>>>((const float4*)part,
                                                            (float4*)out);
}
